// ViT_video_33028298506693
// MI455X (gfx1250) — matrix operand order repaired and checked by reading, not compiled, not run on hardware
//
#include <hip/hip_runtime.h>
#include <hip/hip_bf16.h>
#include <math.h>

// ---------------------------------------------------------------------------
// MI455X (gfx1250, wave32) video-ViT forward.
// GEMMs: v_wmma_f32_16x16x32_f16, block 64x128, 8 waves, wave tile 32x32
// (4 WMMAs/k-step). fp32->f16 conversion during LDS staging; A/B fragments are
// contiguous 8-half runs => ds_load_b128 pairs (B tile transposed in LDS).
// Edge tiles: clamped unconditional loads (batched, then masked) — branch-free.
// Sequence transposes use the async global<->LDS DMA path (ASYNCcnt).
// ---------------------------------------------------------------------------

typedef _Float16 v16h __attribute__((ext_vector_type(16)));
typedef _Float16 v8h  __attribute__((ext_vector_type(8)));
typedef float    v8f  __attribute__((ext_vector_type(8)));

#define BM 64
#define BN 128
#define BKK 32
#define APAD 40   // 32 + 8 halves => 80B rows, 16B aligned, conflict-free
#define BPAD 40

__device__ __forceinline__ float gelu_exact(float v) {
    return 0.5f * v * (1.0f + erff(v * 0.70710678118654752f));
}
__device__ __forceinline__ float wred_sum(float v) {
#pragma unroll
    for (int o = 16; o > 0; o >>= 1) v += __shfl_xor(v, o, 32);
    return v;
}
__device__ __forceinline__ float wred_max(float v) {
#pragma unroll
    for (int o = 16; o > 0; o >>= 1) v = fmaxf(v, __shfl_xor(v, o, 32));
    return v;
}
__device__ __forceinline__ int imin(int a, int b) { return a < b ? a : b; }

// ---------------------------------------------------------------------------
// Generic batched WMMA GEMM:  C[z] = epilogue( scale * A[z] x B[z] )
//   flags bit0: B is [N,K] (transposed), bit1: GELU, bit2: add residual Res.
// ---------------------------------------------------------------------------
__global__ __launch_bounds__(256) void k_gemm(
    const float* __restrict__ A, const float* __restrict__ Bm,
    const float* __restrict__ bias, const float* __restrict__ Res,
    float* __restrict__ C,
    int M, int N, int K, int lda, int ldb, int ldc,
    long long sAo, long long sAi, long long sBo, long long sBi,
    long long sCo, long long sCi,
    int innerB, float scale, int flags) {
    __shared__ _Float16 As[BM][APAD];     // [m][k]
    __shared__ _Float16 Bs[BN][BPAD];     // [n][k]  (transposed tile)

    const int zo = blockIdx.z / innerB, zi = blockIdx.z % innerB;
    A  += (long long)zo * sAo + (long long)zi * sAi;
    Bm += (long long)zo * sBo + (long long)zi * sBi;
    float* Cp = C + (long long)zo * sCo + (long long)zi * sCi;
    const float* R = (flags & 4) ? (Res + (long long)zo * sCo + (long long)zi * sCi) : nullptr;

    const int m0 = blockIdx.y * BM, n0 = blockIdx.x * BN;
    const int tid = threadIdx.x, lane = tid & 31, w = tid >> 5;
    const int wm = w & 1, wn = w >> 1;
    const bool transB = (flags & 1);
    const bool fullM = (m0 + BM) <= M;
    const bool fullN = (n0 + BN) <= N;

    const int arow = tid >> 2, akk = (tid & 3) * 8;        // A: 8 halves/thread
    const int bnn = tid >> 1, bkk = (tid & 1) * 16;        // B: 16 halves/thread

    // clamped (always in-bounds) staging bases for edge tiles
    const float aMask = ((m0 + arow) < M) ? 1.0f : 0.0f;
    const int   aRow  = imin(m0 + arow, M - 1);
    const float bMask = ((n0 + bnn) < N) ? 1.0f : 0.0f;
    const int   bRow  = imin(n0 + bnn, N - 1);

    v8f acc[2][2];
#pragma unroll
    for (int sm = 0; sm < 2; sm++)
#pragma unroll
        for (int sn = 0; sn < 2; sn++)
#pragma unroll
            for (int r = 0; r < 8; r++) acc[sm][sn][r] = 0.0f;

    for (int k0 = 0; k0 < K; k0 += BKK) {
        const bool fullK = (k0 + BKK) <= K;
        // ---- stage A tile 64x32 ----
        {
            v8h hv;
            if (fullM && fullK) {
                const float* ap = A + (long long)(m0 + arow) * lda + k0 + akk;
                __builtin_prefetch((const void*)(ap + BKK), 0, 1);
#pragma unroll
                for (int j = 0; j < 8; j++) hv[j] = (_Float16)ap[j];
            } else {   // branch-free: batch clamped loads, then mask-convert
                const float* ap = A + (long long)aRow * lda;
                float vals[8];
#pragma unroll
                for (int j = 0; j < 8; j++)
                    vals[j] = ap[imin(k0 + akk + j, K - 1)];
#pragma unroll
                for (int j = 0; j < 8; j++) {
                    const float m = ((k0 + akk + j) < K) ? aMask : 0.0f;
                    hv[j] = (_Float16)(vals[j] * m);
                }
            }
            *(v8h*)&As[arow][akk] = hv;
        }
        // ---- stage B tile 32x128 into transposed LDS ----
        {
            v8h h0, h1;
            if (!transB) {  // B[K,N] row-major: strided reads, contiguous store
                if (fullN && fullK) {
                    const float* bp = Bm + (long long)(k0 + bkk) * ldb + n0 + bnn;
                    __builtin_prefetch((const void*)(bp + (long long)BKK * ldb), 0, 1);
#pragma unroll
                    for (int j = 0; j < 8; j++) h0[j] = (_Float16)bp[(long long)j * ldb];
#pragma unroll
                    for (int j = 0; j < 8; j++) h1[j] = (_Float16)bp[(long long)(j + 8) * ldb];
                } else {
                    const float* bp = Bm + bRow;
                    float vals[16];
#pragma unroll
                    for (int j = 0; j < 16; j++)
                        vals[j] = bp[(long long)imin(k0 + bkk + j, K - 1) * ldb];
#pragma unroll
                    for (int j = 0; j < 16; j++) {
                        const float m = ((k0 + bkk + j) < K) ? bMask : 0.0f;
                        const _Float16 hvv = (_Float16)(vals[j] * m);
                        if (j < 8) h0[j] = hvv; else h1[j - 8] = hvv;
                    }
                }
            } else {        // B[N,K] row-major: contiguous reads
                if (fullN && fullK) {
                    const float* bp = Bm + (long long)(n0 + bnn) * ldb + k0 + bkk;
                    __builtin_prefetch((const void*)(bp + BKK), 0, 1);
#pragma unroll
                    for (int j = 0; j < 8; j++) h0[j] = (_Float16)bp[j];
#pragma unroll
                    for (int j = 0; j < 8; j++) h1[j] = (_Float16)bp[j + 8];
                } else {
                    const float* bp = Bm + (long long)bRow * ldb;
                    float vals[16];
#pragma unroll
                    for (int j = 0; j < 16; j++)
                        vals[j] = bp[imin(k0 + bkk + j, K - 1)];
#pragma unroll
                    for (int j = 0; j < 16; j++) {
                        const float m = ((k0 + bkk + j) < K) ? bMask : 0.0f;
                        const _Float16 hvv = (_Float16)(vals[j] * m);
                        if (j < 8) h0[j] = hvv; else h1[j - 8] = hvv;
                    }
                }
            }
            *(v8h*)&Bs[bnn][bkk] = h0;
            *(v8h*)&Bs[bnn][bkk + 8] = h1;
        }
        __syncthreads();

        // ---- fragments (contiguous k = 16*(lane/16) + s, same for A and B) + 4 WMMAs ----
        const int ar = wm * 32 + (lane & 15);
        const int kbA = (lane < 16) ? 0 : 16;
        const int kbB = (lane < 16) ? 0 : 16;

        v8h a0lo = *(const v8h*)&As[ar][kbA];
        v8h a0hi = *(const v8h*)&As[ar][kbA + 8];
        v8h a1lo = *(const v8h*)&As[ar + 16][kbA];
        v8h a1hi = *(const v8h*)&As[ar + 16][kbA + 8];
        v16h af0 = __builtin_shufflevector(a0lo, a0hi, 0,1,2,3,4,5,6,7,8,9,10,11,12,13,14,15);
        v16h af1 = __builtin_shufflevector(a1lo, a1hi, 0,1,2,3,4,5,6,7,8,9,10,11,12,13,14,15);

        const int bc0 = wn * 32 + (lane & 15);
        v8h b0lo = *(const v8h*)&Bs[bc0][kbB];
        v8h b0hi = *(const v8h*)&Bs[bc0][kbB + 8];
        v8h b1lo = *(const v8h*)&Bs[bc0 + 16][kbB];
        v8h b1hi = *(const v8h*)&Bs[bc0 + 16][kbB + 8];
        v16h bf0 = __builtin_shufflevector(b0lo, b0hi, 0,1,2,3,4,5,6,7,8,9,10,11,12,13,14,15);
        v16h bf1 = __builtin_shufflevector(b1lo, b1hi, 0,1,2,3,4,5,6,7,8,9,10,11,12,13,14,15);

        acc[0][0] = __builtin_amdgcn_wmma_f32_16x16x32_f16(false, af0, false, bf0,
                                                           (short)0, acc[0][0], false, false);
        acc[0][1] = __builtin_amdgcn_wmma_f32_16x16x32_f16(false, af0, false, bf1,
                                                           (short)0, acc[0][1], false, false);
        acc[1][0] = __builtin_amdgcn_wmma_f32_16x16x32_f16(false, af1, false, bf0,
                                                           (short)0, acc[1][0], false, false);
        acc[1][1] = __builtin_amdgcn_wmma_f32_16x16x32_f16(false, af1, false, bf1,
                                                           (short)0, acc[1][1], false, false);
        __syncthreads();
    }

    // ---- epilogue. C layout: VGPR r -> M = r (+8 lanes>=16), N = lane&15 ----
    const int rbase = m0 + wm * 32 + ((lane < 16) ? 0 : 8);
#pragma unroll
    for (int sn = 0; sn < 2; sn++) {
        const int gn = n0 + wn * 32 + sn * 16 + (lane & 15);
        if (gn < N) {
#pragma unroll
            for (int sm = 0; sm < 2; sm++) {
#pragma unroll
                for (int r = 0; r < 8; r++) {
                    const int gm = rbase + sm * 16 + r;
                    if (gm < M) {
                        float v = acc[sm][sn][r] * scale;
                        if (bias) v += bias[gn];
                        if (flags & 2) v = gelu_exact(v);
                        const long long co = (long long)gm * ldc + gn;
                        if (flags & 4) v += R[co];
                        Cp[co] = v;
                    }
                }
            }
        }
    }
}

// ---------------------------------------------------------------------------
// LayerNorm over D=768, one wave per row, row remap on input and output.
// ---------------------------------------------------------------------------
__global__ __launch_bounds__(256) void k_layernorm(
    const float* __restrict__ X, float* __restrict__ Y,
    const float* __restrict__ g, const float* __restrict__ b, float eps,
    int rows, int innerIn, long long sInO, long long sInI,
    int innerOut, long long sOutO, long long sOutI) {
    const int row = blockIdx.x * 8 + (threadIdx.x >> 5);
    if (row >= rows) return;
    const int lane = threadIdx.x & 31;
    const float* xin = X + (long long)(row / innerIn) * sInO + (long long)(row % innerIn) * sInI;
    float* yo = Y + (long long)(row / innerOut) * sOutO + (long long)(row % innerOut) * sOutI;
    float s = 0.0f, s2 = 0.0f;
#pragma unroll
    for (int j = 0; j < 24; j++) {
        float v = xin[lane + 32 * j];
        s += v; s2 += v * v;
    }
    s = wred_sum(s); s2 = wred_sum(s2);
    const float mean = s * (1.0f / 768.0f);
    const float var = s2 * (1.0f / 768.0f) - mean * mean;
    const float inv = rsqrtf(var + eps);
#pragma unroll
    for (int j = 0; j < 24; j++) {
        const int i = lane + 32 * j;
        yo[i] = (xin[i] - mean) * inv * g[i] + b[i];
    }
}

// Row softmax, one wave per row, in place.
__global__ __launch_bounds__(256) void k_softmax(float* __restrict__ X, int rows, int L) {
    const int row = blockIdx.x * 8 + (threadIdx.x >> 5);
    if (row >= rows) return;
    const int lane = threadIdx.x & 31;
    float* p = X + (long long)row * L;
    float mx = -3.4e38f;
    for (int i = lane; i < L; i += 32) mx = fmaxf(mx, p[i]);
    mx = wred_max(mx);
    float s = 0.0f;
    for (int i = lane; i < L; i += 32) { float e = expf(p[i] - mx); p[i] = e; s += e; }
    s = wred_sum(s);
    const float inv = 1.0f / s;
    for (int i = lane; i < L; i += 32) p[i] *= inv;
}

// ---------------------------------------------------------------------------
// STFT: separable DFT over (win=5, H=224, W=224) per (b,c,t).
// ---------------------------------------------------------------------------
__global__ void k_dft_win(const float* __restrict__ X, float2* __restrict__ O, long long n) {
    long long idx = (long long)blockIdx.x * blockDim.x + threadIdx.x;
    if (idx >= n) return;
    long long r = idx;
    const int wq = r % 224; r /= 224;
    const int h = r % 224; r /= 224;
    const int j = r % 5; r /= 5;
    const int t = r % 16; r /= 16;
    const int c = r % 3;
    const int b = (int)(r / 3);
    float re = 0.0f, im = 0.0f;
#pragma unroll
    for (int mm = 0; mm < 5; mm++) {
        int tt = t + mm - 2;
        if (tt < 0) tt = -tt;
        if (tt > 15) tt = 30 - tt;
        const float v = X[((((long long)(b * 3 + c) * 16 + tt) * 224 + h) * 224) + wq];
        const float ang = -6.2831853071795864f * (float)((j * mm) % 5) * 0.2f;
        float sn, cs; __sincosf(ang, &sn, &cs);
        re += v * cs; im += v * sn;
    }
    float2 o; o.x = re; o.y = im; O[idx] = o;
}

__global__ void k_dft_line(const float2* __restrict__ I, float2* __restrict__ O,
                           int axis, long long n) {
    long long idx = (long long)blockIdx.x * blockDim.x + threadIdx.x;
    if (idx >= n) return;
    const long long plane = idx / 50176;
    const int ip = (int)(idx % 50176);
    int kidx, fixed;
    if (axis == 0) { kidx = ip / 224; fixed = ip % 224; }
    else           { kidx = ip % 224; fixed = ip / 224; }
    const float step = -6.2831853071795864f / 224.0f;
    float re = 0.0f, im = 0.0f;
    for (int nIt = 0; nIt < 224; nIt++) {
        const long long src = plane * 50176 +
            (axis == 0 ? (long long)nIt * 224 + fixed : (long long)fixed * 224 + nIt);
        const float2 u = I[src];
        const float ang = step * (float)((kidx * nIt) % 224);
        float sn, cs; __sincosf(ang, &sn, &cs);
        re += u.x * cs - u.y * sn;
        im += u.x * sn + u.y * cs;
    }
    float2 o; o.x = re; o.y = im; O[idx] = o;
}

__global__ void k_fft_im2col(const float2* __restrict__ F, float* __restrict__ A, long long n) {
    long long idx = (long long)blockIdx.x * blockDim.x + threadIdx.x;
    if (idx >= n) return;
    const int col = (int)(idx % 3840);
    const long long row = idx / 3840;
    const int p = (int)(row % 196);
    const int bt = (int)(row / 196);
    const int b = bt / 16, t = bt % 16;
    const int py = p / 14, px = p % 14;
    const int j = col / 768;
    const int rem = col % 768;
    const int c = rem >> 8, r2 = rem & 255, ky = r2 >> 4, kx = r2 & 15;
    const int hs = (py * 16 + ky + 112) % 224;
    const int wsx = (px * 16 + kx + 112) % 224;
    const int js = (j + 3) % 5;
    const float2 u = F[((((long long)(b * 3 + c) * 16 + t) * 5 + js) * 224 + hs) * 224 + wsx];
    A[idx] = log1pf(sqrtf(u.x * u.x + u.y * u.y));
}

__global__ void k_im2col_patch(const float* __restrict__ X, float* __restrict__ A, long long n) {
    long long idx = (long long)blockIdx.x * blockDim.x + threadIdx.x;
    if (idx >= n) return;
    const int col = (int)(idx % 768);
    const long long row = idx / 768;
    const int p = (int)(row % 196);
    const int bt = (int)(row / 196);
    const int b = bt / 16, t = bt % 16;
    const int py = p / 14, px = p % 14;
    const int c = col >> 8, r2 = col & 255, ky = r2 >> 4, kx = r2 & 15;
    A[idx] = X[((((long long)(b * 3 + c) * 16 + t) * 224 + py * 16 + ky) * 224) + px * 16 + kx];
}

__global__ void k_assemble(const float* __restrict__ cls, const float* __restrict__ pos,
                           const float* __restrict__ temb, const float* __restrict__ xe,
                           const float* __restrict__ xt, const float* __restrict__ xm,
                           float* __restrict__ H, long long n) {
    long long idx = (long long)blockIdx.x * blockDim.x + threadIdx.x;
    if (idx >= n) return;
    const int d = (int)(idx % 768);
    const long long row = idx / 768;
    const int j = (int)(row % 197);
    const int bt = (int)(row / 197);
    const int t = bt % 16;
    float v;
    if (j == 0) v = cls[d];
    else {
        const long long q = ((long long)bt * 196 + (j - 1)) * 768 + d;
        v = xe[q] + xt[q] + xm[q];
    }
    H[idx] = v + pos[(long long)j * 768 + d] + temb[(long long)t * 768 + d];
}

__global__ void k_addpost(float* __restrict__ H, const float* __restrict__ xt,
                          const float* __restrict__ xm, long long n) {
    long long idx = (long long)blockIdx.x * blockDim.x + threadIdx.x;
    if (idx >= n) return;
    const int d = (int)(idx % 768);
    const long long row = idx / 768;
    const int j = (int)(row % 196);
    const int bt = (int)(row / 196);
    H[((long long)bt * 197 + 1 + j) * 768 + d] += xt[idx] + xm[idx];
}

// ---------------------------------------------------------------------------
// Sequence transpose as an async global->LDS->global DMA row copy (CDNA5 path).
// mode 0: dst[(b*196+n)*16+t] = src[(b*16+t)*196+n]   (rows of 768 floats)
// mode 1: dst[(b*16+t)*196+n] = src[(b*196+n)*16+t]
// ---------------------------------------------------------------------------
__global__ __launch_bounds__(256) void k_seq_async(const float* __restrict__ S,
                                                   float* __restrict__ D, int mode) {
    __shared__ float lbuf[8][768];
    const int wv = threadIdx.x >> 5, lane = threadIdx.x & 31;
    const int r = blockIdx.x * 8 + wv;   // destination row, 0..6271
    int srcRow;
    if (mode == 0) {
        const int t = r & 15; const int bn = r >> 4;
        const int nn = bn % 196; const int b = bn / 196;
        srcRow = (b * 16 + t) * 196 + nn;
    } else {
        const int nn = r % 196; const int bt = r / 196;
        const int t = bt & 15; const int b = bt >> 4;
        srcRow = (b * 196 + nn) * 16 + t;
    }
    const float* sp = S + (long long)srcRow * 768;
    float* dp = D + (long long)r * 768;
    const unsigned lbase = (unsigned)(size_t)(&lbuf[wv][0]);   // LDS offset (addr[31:0])
#pragma unroll
    for (int i = 0; i < 6; i++) {
        const unsigned off = (unsigned)(lane * 16 + i * 512);
        const unsigned loff = lbase + off;
        const unsigned long long ga = (unsigned long long)(size_t)sp + off;
        asm volatile("global_load_async_to_lds_b128 %0, %1, off"
                     :: "v"(loff), "v"(ga) : "memory");
    }
    asm volatile("s_wait_asynccnt 0x0" ::: "memory");
#pragma unroll
    for (int i = 0; i < 6; i++) {
        const unsigned off = (unsigned)(lane * 16 + i * 512);
        const unsigned loff = lbase + off;
        const unsigned long long ga = (unsigned long long)(size_t)dp + off;
        asm volatile("global_store_async_from_lds_b128 %0, %1, off"
                     :: "v"(ga), "v"(loff) : "memory");
    }
    asm volatile("s_wait_asynccnt 0x0" ::: "memory");
}

// ---------------------------------------------------------------------------
// Host orchestration
// ---------------------------------------------------------------------------
extern "C" void kernel_launch(void* const* d_in, const int* in_sizes, int n_in,
                              void* d_out, int out_size, void* d_ws, size_t ws_size,
                              hipStream_t stream) {
    (void)in_sizes; (void)n_in; (void)out_size; (void)ws_size;
    int ii = 0;
    auto nxt = [&]() { return (const float*)d_in[ii++]; };

    const float* X = nxt();
    const float* patch_w = nxt(); const float* patch_b = nxt();
    const float* fft_w = nxt();   const float* fft_b = nxt();
    const float* cls = nxt(); const float* pos = nxt(); const float* temb = nxt();
    const float* fnorm_w = nxt(); const float* fnorm_b = nxt();

    struct Blk { const float *ln1w,*ln1b,*qkvw,*qkvb,*projw,*projb,*ln2w,*ln2b,*fc1w,*fc1b,*fc2w,*fc2b; } BL[12];
    for (int i = 0; i < 12; i++) {
        BL[i].ln1w=nxt(); BL[i].ln1b=nxt(); BL[i].qkvw=nxt(); BL[i].qkvb=nxt();
        BL[i].projw=nxt(); BL[i].projb=nxt(); BL[i].ln2w=nxt(); BL[i].ln2b=nxt();
        BL[i].fc1w=nxt(); BL[i].fc1b=nxt(); BL[i].fc2w=nxt(); BL[i].fc2b=nxt();
    }
    struct Pr { const float *tfc1w,*tfc1b,*tqkvw,*tprojw,*tprojb,*tfc2w,*tfc2b,
                            *normw,*normb,*sfc1w,*sfc1b,*sfc2w,*sfc2b,
                            *d1w,*d1b,*d2w,*d2b,*d3w,*d3b; } PR[12];
    for (int i = 0; i < 12; i++) {
        PR[i].tfc1w=nxt(); PR[i].tfc1b=nxt(); PR[i].tqkvw=nxt(); PR[i].tprojw=nxt();
        PR[i].tprojb=nxt(); PR[i].tfc2w=nxt(); PR[i].tfc2b=nxt(); PR[i].normw=nxt();
        PR[i].normb=nxt(); PR[i].sfc1w=nxt(); PR[i].sfc1b=nxt(); PR[i].sfc2w=nxt();
        PR[i].sfc2b=nxt(); PR[i].d1w=nxt(); PR[i].d1b=nxt(); PR[i].d2w=nxt();
        PR[i].d2b=nxt(); PR[i].d3w=nxt(); PR[i].d3b=nxt();
    }
    const float *pxnw[12], *pxnb[12], *pfnw[12], *pfnb[12];
    for (int i = 0; i < 12; i++) { pxnw[i]=nxt(); pxnb[i]=nxt(); }
    for (int i = 0; i < 12; i++) { pfnw[i]=nxt(); pfnb[i]=nxt(); }

    // ---- workspace layout (floats) ----
    constexpr long long NCPX = 24084480LL;   // 2*3*16*5*224*224 complex elems
    constexpr long long W0F  = 2 * NCPX;
    constexpr long long SZ196 = 4816896LL;   // 32*196*768
    constexpr long long SZ197 = 4841472LL;   // 32*197*768
    float* wsf = (float*)d_ws;
    float* W0 = wsf;
    float* W1 = wsf + W0F;
    float* xe  = wsf + 2 * W0F;
    float* xfE = xe + SZ196;
    float* xm  = xfE + SZ196;
    float* xt  = xm + SZ196;
    float* xsN = xt + SZ196;
    float* xfN = xsN + SZ196;
    float* hF  = xfN + SZ196;
    float* aLn = hF + SZ197;
    float* qkv   = W0;                 // 32*197*2304
    float* probs = W0 + 14600000LL;    // up to 32*12*197*197
    float* Afft   = W1;                // 6272*3840
    float* Apatch = W1 + 24084480LL;   // 6272*768
    float* ctx    = W1;                // 32*197*768
    float* xtSeq  = W1 +  5000000LL;   // 392*16*768
    float* tQkv   = W1 + 10000000LL;   // 6272*576
    float* tH     = W1 + 14000000LL;   // 6272*192
    float* tAtt   = W1 + 15500000LL;   // 6272*192
    float* tProj  = W1 + 17000000LL;   // 6272*192
    float* pQ     = W1 + 18500000LL;   // 6272*192
    float* pK     = W1 + 20000000LL;   // 6272*192
    float* xmCtx  = W1 + 21500000LL;   // 6272*192
    float* ff     = W1 + 23000000LL;   // 6304*3072
    float* outp = (float*)d_out;

    auto gemm = [&](const float* A, const float* Bmat, const float* bias, const float* Res,
                    float* C, int M, int N, int K, int lda, int ldb, int ldc,
                    long long sAo, long long sAi, long long sBo, long long sBi,
                    long long sCo, long long sCi, int outer, int inner,
                    float scale, int flags) {
        dim3 g((N + BN - 1) / BN, (M + BM - 1) / BM, outer * inner);
        k_gemm<<<g, 256, 0, stream>>>(A, Bmat, bias, Res, C, M, N, K, lda, ldb, ldc,
                                      sAo, sAi, sBo, sBi, sCo, sCi, inner, scale, flags);
    };
    auto gemmW = [&](const float* A, const float* Wm, const float* bias, const float* Res,
                     float* C, int M, int N, int K, int flags) {
        int ldb = (flags & 1) ? K : N;
        gemm(A, Wm, bias, Res, C, M, N, K, K, ldb, N, 0, 0, 0, 0, 0, 0, 1, 1, 1.0f, flags);
    };
    auto lnrun = [&](const float* Xi, float* Y, const float* g, const float* b, float eps,
                     int rows, int innerIn, long long sInO, long long sInI,
                     int innerOut, long long sOutO, long long sOutI) {
        k_layernorm<<<(rows + 7) / 8, 256, 0, stream>>>(Xi, Y, g, b, eps, rows,
                                                        innerIn, sInO, sInI,
                                                        innerOut, sOutO, sOutI);
    };
    auto lnC = [&](const float* Xi, float* Y, const float* g, const float* b,
                   float eps, int rows) {
        lnrun(Xi, Y, g, b, eps, rows, rows, 0, 768, rows, 0, 768);
    };
    auto smx = [&](float* Xi, int rows, int L) {
        k_softmax<<<(rows + 7) / 8, 256, 0, stream>>>(Xi, rows, L);
    };
    auto blocks1d = [](long long n) { return dim3((unsigned)((n + 255) / 256)); };

    // ---- STFT ----
    k_dft_win<<<blocks1d(NCPX), 256, 0, stream>>>(X, (float2*)W0, NCPX);
    k_dft_line<<<blocks1d(NCPX), 256, 0, stream>>>((const float2*)W0, (float2*)W1, 0, NCPX);
    k_dft_line<<<blocks1d(NCPX), 256, 0, stream>>>((const float2*)W1, (float2*)W0, 1, NCPX);
    k_fft_im2col<<<blocks1d(6272LL * 3840), 256, 0, stream>>>((const float2*)W0, Afft, 6272LL * 3840);
    k_im2col_patch<<<blocks1d(6272LL * 768), 256, 0, stream>>>(X, Apatch, 6272LL * 768);

    // ---- patch embeddings ----
    gemmW(Apatch, patch_w, patch_b, nullptr, xe, 6272, 768, 768, /*transB*/1);
    gemmW(Afft, fft_w, fft_b, nullptr, xfE, 6272, 768, 3840, /*transB*/1);

    // ---- prompt block ----
    auto promptBlock = [&](int pi, const float* xs, const float* xfn,
                           float* xtOut, float* xmOut) {
        const Pr& p = PR[pi];
        k_seq_async<<<784, 256, 0, stream>>>(xs, xtSeq, 0);
        gemmW(xtSeq, p.tfc1w, p.tfc1b, nullptr, tH, 6272, 192, 768, /*gelu*/2);
        gemmW(tH, p.tqkvw, nullptr, nullptr, tQkv, 6272, 576, 192, 0);
        gemm(tQkv, tQkv + 192, nullptr, nullptr, probs, 16, 16, 24, 576, 576, 16,
             16LL * 576, 24, 16LL * 576, 24, 8LL * 256, 256, 392, 8,
             0.20412414523193154f, /*transB*/1);
        smx(probs, 392 * 8 * 16, 16);
        gemm(probs, tQkv + 384, nullptr, nullptr, tAtt, 16, 24, 16, 16, 576, 192,
             8LL * 256, 256, 16LL * 576, 24, 16LL * 192, 24, 392, 8, 1.0f, 0);
        gemmW(tAtt, p.tprojw, p.tprojb, nullptr, tProj, 6272, 192, 192, 0);
        gemmW(tProj, p.tfc2w, p.tfc2b, nullptr, xtSeq, 6272, 768, 192, 0);
        k_seq_async<<<784, 256, 0, stream>>>(xtSeq, xtOut, 1);
        lnC(xtOut, xtSeq, p.normw, p.normb, 1e-5f, 6272);
        gemmW(xtSeq, p.sfc1w, p.sfc1b, nullptr, tH, 6272, 192, 768, /*gelu*/2);
        gemmW(tH, p.sfc2w, p.sfc2b, nullptr, xtOut, 6272, 768, 192, 0);
        gemmW(xs, p.d1w, p.d1b, nullptr, pQ, 6272, 192, 768, 0);
        gemmW(xfn, p.d2w, p.d2b, nullptr, pK, 6272, 192, 768, 0);
        gemm(pQ, pK, nullptr, nullptr, probs, 196, 196, 192, 192, 192, 196,
             196LL * 192, 0, 196LL * 192, 0, 196LL * 196, 0, 32, 1, 1.0f, /*transB*/1);
        smx(probs, 32 * 196, 196);
        gemm(probs, pK, nullptr, nullptr, xmCtx, 196, 192, 196, 196, 192, 192,
             196LL * 196, 0, 196LL * 192, 0, 196LL * 192, 0, 32, 1, 1.0f, 0);
        gemmW(xmCtx, p.d3w, p.d3b, nullptr, xmOut, 6272, 768, 192, 0);
    };

    // ---- layer 0 prompt + token assembly ----
    lnC(xe, xsN, pxnw[0], pxnb[0], 1e-5f, 6272);
    lnC(xfE, xfN, pfnw[0], pfnb[0], 1e-5f, 6272);
    promptBlock(0, xsN, xfN, xt, xm);
    k_assemble<<<blocks1d(SZ197), 256, 0, stream>>>(cls, pos, temb, xe, xt, xm, hF, SZ197);

    // ---- 12 transformer blocks ----
    for (int i = 0; i < 12; i++) {
        if (i >= 1) {
            lnrun(hF + 768, xsN, pxnw[i], pxnb[i], 1e-5f, 6272,
                  196, 197LL * 768, 768, 6272, 0, 768);
            lnC(xm, xfN, pfnw[i], pfnb[i], 1e-5f, 6272);
            promptBlock(i, xsN, xfN, xt, xm);
            k_addpost<<<blocks1d(SZ196), 256, 0, stream>>>(hF, xt, xm, SZ196);
        }
        const Blk& bk = BL[i];
        lnC(hF, aLn, bk.ln1w, bk.ln1b, 1e-6f, 6304);
        gemmW(aLn, bk.qkvw, bk.qkvb, nullptr, qkv, 6304, 2304, 768, 0);
        gemm(qkv, qkv + 768, nullptr, nullptr, probs, 197, 197, 64, 2304, 2304, 197,
             197LL * 2304, 64, 197LL * 2304, 64, 12LL * 38809, 38809, 32, 12,
             0.125f, /*transB*/1);
        smx(probs, 32 * 12 * 197, 197);
        gemm(probs, qkv + 1536, nullptr, nullptr, ctx, 197, 64, 197, 197, 2304, 768,
             12LL * 38809, 38809, 197LL * 2304, 64, 197LL * 768, 64, 32, 12, 1.0f, 0);
        gemmW(ctx, bk.projw, bk.projb, hF, hF, 6304, 768, 768, /*residual*/4);
        lnC(hF, aLn, bk.ln2w, bk.ln2b, 1e-6f, 6304);
        gemmW(aLn, bk.fc1w, bk.fc1b, nullptr, ff, 6304, 3072, 768, /*gelu*/2);
        gemmW(ff, bk.fc2w, bk.fc2b, hF, hF, 6304, 768, 3072, /*residual*/4);
    }

    // ---- final LN on cls rows only, straight to d_out (B,T,D) ----
    lnrun(hF, outp, fnorm_w, fnorm_b, 1e-6f, 32,
          32, 0, 197LL * 768, 32, 0, 768);
}
